// AxialAttention_4578435137973
// MI455X (gfx1250) — matrix operand order repaired and checked by reading, not compiled, not run on hardware
//
#include <hip/hip_runtime.h>

// ---------------------------------------------------------------------------
// AxialAttention (full-grid attention, group-of-32 softmax) for MI455X gfx1250.
// All matmuls on v_wmma_f32_16x16x32_bf16 (wave32).
// - Weights pre-transposed in the bf16 conversion pass; V stored transposed by
//   the QKV epilogue => ALL LDS staging is global_load_async_to_lds_b128.
// - Double-buffered (ping-pong) LDS tiles: tile i+1 DMA overlaps WMMA on i.
// - Every WMMA fragment read is a contiguous ds_load_b128 pair.
//
// Workspace (~44 MB): Xbf, Wqt, Wot, Qb, Kb, Vt, Ab (all bf16/ushort).
// ---------------------------------------------------------------------------

typedef __attribute__((ext_vector_type(16))) __bf16          v16bf;
typedef __attribute__((ext_vector_type(16))) unsigned short  v16us;
typedef __attribute__((ext_vector_type(8)))  float           v8f;

__device__ __forceinline__ unsigned short f32_to_bf16(float f) {
  unsigned int u = __builtin_bit_cast(unsigned int, f);
  u += 0x7FFFu + ((u >> 16) & 1u);          // round-to-nearest-even
  return (unsigned short)(u >> 16);
}

__device__ __forceinline__ v8f wmma_bf16(v16bf a, v16bf b, v8f c) {
  return __builtin_amdgcn_wmma_f32_16x16x32_bf16(
      /*neg_a=*/false, a, /*neg_b=*/false, b,
      /*c_mod=*/(short)0, c, /*reuse_a=*/false, /*reuse_b=*/false);
}

// Async global->LDS copy of 16 bytes (ASYNCcnt tracked, no VGPR round-trip).
__device__ __forceinline__ void async_copy_b128(const void* gptr, void* lptr) {
  unsigned lds           = (unsigned)(uintptr_t)lptr;
  unsigned long long gad = (unsigned long long)(uintptr_t)gptr;
  asm volatile("global_load_async_to_lds_b128 %0, %1, off"
               :: "v"(lds), "v"(gad) : "memory");
}
__device__ __forceinline__ void wait_async() {
  asm volatile("s_wait_asynccnt 0x0" ::: "memory");
}

// A fragment (16x32 bf16), memory row-major [m][k].
__device__ __forceinline__ v16bf load_frag_a(const unsigned short* p,
                                             int row_base, int ld, int kbase) {
  const int lane = threadIdx.x & 31;
  const int g = lane >> 4;
  const unsigned short* q = p + (size_t)(row_base + (lane & 15)) * ld + kbase + g * 16;
  v16us t;
#pragma unroll
  for (int e = 0; e < 16; ++e) t[e] = q[e];
  return __builtin_bit_cast(v16bf, t);
}

// B fragment (32x16 bf16) from B-transposed memory: mem[n][k] (k contiguous).
__device__ __forceinline__ v16bf load_frag_b_nk(const unsigned short* p,
                                                int n_base, int ld, int kbase) {
  const int lane = threadIdx.x & 31;
  const unsigned short* q =
      p + (size_t)(n_base + (lane & 15)) * ld + kbase + ((lane >> 4) << 4);
  v16us t;
#pragma unroll
  for (int e = 0; e < 16; ++e) t[e] = q[e];
  return __builtin_bit_cast(v16bf, t);
}

// ---------------------------------------------------------------------------
// Kernel 1a: f32 -> bf16 (straight copy)
__global__ void cvt_bf16_kernel(const float* __restrict__ src,
                                unsigned short* __restrict__ dst, int n) {
  int i = blockIdx.x * blockDim.x + threadIdx.x;
  if (i < n) dst[i] = f32_to_bf16(src[i]);
}

// Kernel 1b: f32 [512][cols] -> bf16 transposed [cols][512]
__global__ void cvt_bf16_T512_kernel(const float* __restrict__ src,
                                     unsigned short* __restrict__ dst,
                                     int cols, int total) {
  int i = blockIdx.x * blockDim.x + threadIdx.x;
  if (i < total) {
    int n = i >> 9;             // output row = source column
    int k = i & 511;            // output col = source row
    dst[i] = f32_to_bf16(src[(size_t)k * cols + n]);
  }
}

// ---------------------------------------------------------------------------
// Kernel 2: QKV GEMM [8192,512] @ [512,1536]. Block tile 128x128, 8 waves,
// each 32x64 (2x4 WMMA tiles => 8 wmma / 32-K step). Ping-pong async tiles.
// Scatter: Q/K -> [bh][pos][64] (Q pre-scaled 1/8); V -> [bh][64][pos].
// ---------------------------------------------------------------------------
__global__ __launch_bounds__(256) void gemm_qkv_kernel(
    const unsigned short* __restrict__ X, const unsigned short* __restrict__ Wt,
    unsigned short* __restrict__ Qb, unsigned short* __restrict__ Kb,
    unsigned short* __restrict__ Vt) {
  __shared__ __align__(16) unsigned short As[2][128 * 40];  // [m][k], ld=40
  __shared__ __align__(16) unsigned short Bs[2][128 * 40];  // [n][k], ld=40

  const int tid = threadIdx.x;
  const int lane = tid & 31;
  const int wave = tid >> 5;
  const int m0 = blockIdx.x * 128;
  const int n0 = blockIdx.y * 128;
  const int wm = wave >> 1;   // 0..3 -> 32 rows
  const int wn = wave & 1;    // 0..1 -> 64 cols

  auto issue = [&](int k0, int pb) {
#pragma unroll
    for (int it = 0; it < 2; ++it) {
      int idx = tid + it * 256;
      int r = idx >> 2, c = (idx & 3) * 8;
      async_copy_b128(X  + (size_t)(m0 + r) * 512 + k0 + c, As[pb] + r * 40 + c);
      async_copy_b128(Wt + (size_t)(n0 + r) * 512 + k0 + c, Bs[pb] + r * 40 + c);
    }
  };

  v8f acc[2][4] = {};
  issue(0, 0);

#pragma unroll 2
  for (int k0 = 0; k0 < 512; k0 += 32) {
    const int p = (k0 >> 5) & 1;
    wait_async();                       // tile k0 resident
    __syncthreads();
    if (k0 + 32 < 512) issue(k0 + 32, p ^ 1);   // overlap DMA with WMMA
    if (k0 + 64 < 512)
      __builtin_prefetch(X + (size_t)(m0 + (tid & 127)) * 512 + k0 + 64, 0, 1);

    v16bf a[2], b[4];
#pragma unroll
    for (int t = 0; t < 2; ++t) a[t] = load_frag_a(As[p], wm * 32 + t * 16, 40, 0);
#pragma unroll
    for (int u = 0; u < 4; ++u) b[u] = load_frag_b_nk(Bs[p], wn * 64 + u * 16, 40, 0);
#pragma unroll
    for (int t = 0; t < 2; ++t)
#pragma unroll
      for (int u = 0; u < 4; ++u) acc[t][u] = wmma_bf16(a[t], b[u], acc[t][u]);
  }

  // epilogue: block cols lie within one {q,k,v} slab; head fixed per wave
  const int which = n0 >> 9;                 // 0=q 1=k 2=v (block-uniform)
  const int head = ((n0 & 511) >> 6) + wn;
  const int g = lane >> 4;
  if (which == 2) {
#pragma unroll
    for (int t = 0; t < 2; ++t)
#pragma unroll
      for (int u = 0; u < 4; ++u)
#pragma unroll
        for (int r = 0; r < 8; ++r) {
          int gm = m0 + wm * 32 + t * 16 + r + 8 * g;
          int d  = u * 16 + (lane & 15);
          int bb = gm >> 10, pos = gm & 1023;
          Vt[((size_t)(bb * 8 + head) * 64 + d) * 1024 + pos] =
              f32_to_bf16(acc[t][u][r]);               // transposed layout
        }
  } else {
    unsigned short* dst = (which == 0) ? Qb : Kb;
    const float scale = (which == 0) ? 0.125f : 1.0f;
#pragma unroll
    for (int t = 0; t < 2; ++t)
#pragma unroll
      for (int u = 0; u < 4; ++u)
#pragma unroll
        for (int r = 0; r < 8; ++r) {
          int gm = m0 + wm * 32 + t * 16 + r + 8 * g;
          int d  = u * 16 + (lane & 15);
          int bb = gm >> 10, pos = gm & 1023;
          dst[((size_t)(bb * 8 + head) * 1024 + pos) * 64 + d] =
              f32_to_bf16(acc[t][u][r] * scale);
        }
  }
}

// ---------------------------------------------------------------------------
// Kernel 3: attention. grid (64 bh, 16 q-tiles), 256 threads.
// Ping-pong K/V tiles (async); softmax 4 threads/row with shfl_xor.
// ---------------------------------------------------------------------------
__global__ __launch_bounds__(256) void attn_kernel(
    const unsigned short* __restrict__ Qb, const unsigned short* __restrict__ Kb,
    const unsigned short* __restrict__ Vt, unsigned short* __restrict__ Ab) {
  __shared__ __align__(16) unsigned short Ks[2][32 * 72];  // [key][d], ld=72
  __shared__ __align__(16) unsigned short Vs[2][64 * 40];  // [d][key], ld=40
  __shared__ __align__(16) float          Ss[64 * 36];     // [q][key], ld=36
  __shared__ __align__(16) unsigned short Ps[64 * 40];     // [q][key], ld=40

  const int bh = blockIdx.x;       // b*8+head
  const int qt = blockIdx.y;
  const int tid = threadIdx.x;
  const int lane = tid & 31;
  const int wave = tid >> 5;
  const int q0 = qt * 64;

  const unsigned short* Qh = Qb + (size_t)bh * 1024 * 64;
  const unsigned short* Kh = Kb + (size_t)bh * 1024 * 64;
  const unsigned short* Vh = Vt + (size_t)bh * 64 * 1024;  // [d][pos]

  const int mt = wave >> 1;        // query 16-row tile (S and O phases)
  const int ntS = wave & 1;        // key 16-col tile for S
  const int ntO0 = (wave & 1) * 2; // dh 16-col tiles for O: {0,1} or {2,3}

  auto issueKV = [&](int x, int pb) {
    int r = tid >> 3, c = (tid & 7) * 8;
    async_copy_b128(Kh + (size_t)(x * 32 + r) * 64 + c, Ks[pb] + r * 72 + c);
    int r2 = tid >> 2, c2 = (tid & 3) * 8;
    async_copy_b128(Vh + (size_t)r2 * 1024 + x * 32 + c2, Vs[pb] + r2 * 40 + c2);
  };

  // Q fragments persist in registers across all 32 key groups
  v16bf qfrag[2];
#pragma unroll
  for (int ks = 0; ks < 2; ++ks)
    qfrag[ks] = load_frag_a(Qh, q0 + mt * 16, 64, ks * 32);

  v8f oacc[2] = {};
  issueKV(0, 0);

#pragma unroll 2
  for (int x = 0; x < 32; ++x) {
    const int p = x & 1;
    wait_async();                        // K/V tile x resident
    __syncthreads();
    if (x + 1 < 32) issueKV(x + 1, p ^ 1);   // overlap DMA with compute

    // ---- S = Q @ Kx^T (contraction over d=64, two 32-steps) ----
    v8f sacc = {};
#pragma unroll
    for (int ks = 0; ks < 2; ++ks) {
      v16bf bfrag = load_frag_b_nk(Ks[p], ntS * 16, 72, ks * 32);
      sacc = wmma_bf16(qfrag[ks], bfrag, sacc);
    }
    {
      const int g = lane >> 4;
      const int n = ntS * 16 + (lane & 15);
#pragma unroll
      for (int r = 0; r < 8; ++r)
        Ss[(mt * 16 + r + 8 * g) * 36 + n] = sacc[r];
    }
    __syncthreads();

    // ---- softmax over this 32-key group: 4 threads/row, shfl reduction ----
    {
      const int row = tid >> 2;          // 0..63
      const int cb  = (tid & 3) * 8;     // 8 cols per thread
      float mx = -1e30f;
#pragma unroll
      for (int c = 0; c < 8; ++c) mx = fmaxf(mx, Ss[row * 36 + cb + c]);
      mx = fmaxf(mx, __shfl_xor(mx, 1, 32));
      mx = fmaxf(mx, __shfl_xor(mx, 2, 32));
      float sum = 0.f;
#pragma unroll
      for (int c = 0; c < 8; ++c) sum += __expf(Ss[row * 36 + cb + c] - mx);
      sum += __shfl_xor(sum, 1, 32);
      sum += __shfl_xor(sum, 2, 32);
      float inv = 1.f / sum;
#pragma unroll
      for (int c = 0; c < 8; ++c)
        Ps[row * 40 + cb + c] = f32_to_bf16(__expf(Ss[row * 36 + cb + c] - mx) * inv);
    }
    __syncthreads();

    // ---- O += P @ Vx (contraction over 32 keys, single 32-step) ----
    {
      v16bf pfrag = load_frag_a(Ps, mt * 16, 40, 0);
#pragma unroll
      for (int u = 0; u < 2; ++u) {
        v16bf vfrag = load_frag_b_nk(Vs[p], (ntO0 + u) * 16, 40, 0);
        oacc[u] = wmma_bf16(pfrag, vfrag, oacc[u]);
      }
    }
  }

  // write attention output: Ab[b*1024+q][head*64+d], bf16
  const int b = bh >> 3, head = bh & 7;
  const int g = lane >> 4;
#pragma unroll
  for (int u = 0; u < 2; ++u) {
    int d = (ntO0 + u) * 16 + (lane & 15);
#pragma unroll
    for (int r = 0; r < 8; ++r) {
      int qq = q0 + mt * 16 + r + 8 * g;
      Ab[((size_t)b * 1024 + qq) * 512 + head * 64 + d] = f32_to_bf16(oacc[u][r]);
    }
  }
}

// ---------------------------------------------------------------------------
// Kernel 4: out = Ab[8192,512] @ Wo[512,512] + bias -> f32. Tile 128x128,
// ping-pong async tiles (Wo pre-transposed).
// ---------------------------------------------------------------------------
__global__ __launch_bounds__(256) void gemm_out_kernel(
    const unsigned short* __restrict__ A, const unsigned short* __restrict__ Wt,
    const float* __restrict__ bias, float* __restrict__ out) {
  __shared__ __align__(16) unsigned short As[2][128 * 40];
  __shared__ __align__(16) unsigned short Bs[2][128 * 40];

  const int tid = threadIdx.x;
  const int lane = tid & 31;
  const int wave = tid >> 5;
  const int m0 = blockIdx.x * 128;
  const int n0 = blockIdx.y * 128;
  const int wm = wave >> 1;
  const int wn = wave & 1;

  auto issue = [&](int k0, int pb) {
#pragma unroll
    for (int it = 0; it < 2; ++it) {
      int idx = tid + it * 256;
      int r = idx >> 2, c = (idx & 3) * 8;
      async_copy_b128(A  + (size_t)(m0 + r) * 512 + k0 + c, As[pb] + r * 40 + c);
      async_copy_b128(Wt + (size_t)(n0 + r) * 512 + k0 + c, Bs[pb] + r * 40 + c);
    }
  };

  v8f acc[2][4] = {};
  issue(0, 0);

#pragma unroll 2
  for (int k0 = 0; k0 < 512; k0 += 32) {
    const int p = (k0 >> 5) & 1;
    wait_async();
    __syncthreads();
    if (k0 + 32 < 512) issue(k0 + 32, p ^ 1);
    if (k0 + 64 < 512)
      __builtin_prefetch(A + (size_t)(m0 + (tid & 127)) * 512 + k0 + 64, 0, 1);

    v16bf a[2], b[4];
#pragma unroll
    for (int t = 0; t < 2; ++t) a[t] = load_frag_a(As[p], wm * 32 + t * 16, 40, 0);
#pragma unroll
    for (int u = 0; u < 4; ++u) b[u] = load_frag_b_nk(Bs[p], wn * 64 + u * 16, 40, 0);
#pragma unroll
    for (int t = 0; t < 2; ++t)
#pragma unroll
      for (int u = 0; u < 4; ++u) acc[t][u] = wmma_bf16(a[t], b[u], acc[t][u]);
  }

  const int g = lane >> 4;
#pragma unroll
  for (int t = 0; t < 2; ++t)
#pragma unroll
    for (int u = 0; u < 4; ++u)
#pragma unroll
      for (int r = 0; r < 8; ++r) {
        int gm = m0 + wm * 32 + t * 16 + r + 8 * g;
        int gn = n0 + wn * 64 + u * 16 + (lane & 15);
        out[(size_t)gm * 512 + gn] = acc[t][u][r] + bias[gn];
      }
}

// ---------------------------------------------------------------------------
extern "C" void kernel_launch(void* const* d_in, const int* in_sizes, int n_in,
                              void* d_out, int out_size, void* d_ws,
                              size_t ws_size, hipStream_t stream) {
  (void)in_sizes; (void)n_in; (void)out_size; (void)ws_size;

  const float* x     = (const float*)d_in[0];
  const float* w_qkv = (const float*)d_in[1];
  const float* w_out = (const float*)d_in[2];
  const float* b_out = (const float*)d_in[3];

  unsigned short* Xbf = (unsigned short*)d_ws;
  unsigned short* Wqt = Xbf + (size_t)8192 * 512;          // [1536][512]
  unsigned short* Wot = Wqt + (size_t)1536 * 512;          // [512][512] (T)
  unsigned short* Qb  = Wot + (size_t)512 * 512;           // [bh][pos][64]
  unsigned short* Kb  = Qb  + (size_t)64 * 1024 * 64;      // [bh][pos][64]
  unsigned short* Vt  = Kb  + (size_t)64 * 1024 * 64;      // [bh][64][pos]
  unsigned short* Ab  = Vt  + (size_t)64 * 1024 * 64;      // [8192][512]

  cvt_bf16_kernel<<<(8192 * 512 + 255) / 256, 256, 0, stream>>>(x, Xbf, 8192 * 512);
  cvt_bf16_T512_kernel<<<(1536 * 512 + 255) / 256, 256, 0, stream>>>(w_qkv, Wqt,
                                                                     1536, 1536 * 512);
  cvt_bf16_T512_kernel<<<(512 * 512 + 255) / 256, 256, 0, stream>>>(w_out, Wot,
                                                                    512, 512 * 512);

  gemm_qkv_kernel<<<dim3(64, 12), 256, 0, stream>>>(Xbf, Wqt, Qb, Kb, Vt);
  attn_kernel<<<dim3(64, 16), 256, 0, stream>>>(Qb, Kb, Vt, Ab);
  gemm_out_kernel<<<dim3(64, 4), 256, 0, stream>>>(Ab, Wot, b_out, (float*)d_out);
}
